// OriginalAILayerNorm_19765439496652
// MI455X (gfx1250) — compile-verified, matrix operand order checked by reading, not run on hardware
//
#include <hip/hip_runtime.h>

typedef int v8i __attribute__((ext_vector_type(8)));

// INV_SQRT_LUT / 2^16 applied at use site
__constant__ float c_inv_sqrt_lut[16] = {
    65535.f, 46341.f, 32768.f, 23170.f, 16384.f, 11585.f, 8192.f, 5793.f,
    4096.f,  2896.f,  2048.f,  1448.f,  1024.f,  724.f,   512.f,  362.f};

__device__ __forceinline__ float scale_from_bits(unsigned bits) {
    return fmaxf(__uint_as_float(bits) * (1.0f / 127.0f), 1e-8f);
}

// ---------------- init: zero absmax scalars + per-row int accumulators ----
__global__ void k_init(unsigned* w, int n) {
    int i = blockIdx.x * blockDim.x + threadIdx.x;
    if (i < n) w[i] = 0u;
}

// ---------------- pass 1: global max |x| ---------------------------------
__global__ __launch_bounds__(256) void k_absmax(const float4* __restrict__ x,
                                                int n4, unsigned* __restrict__ outBits) {
    const int stride = gridDim.x * blockDim.x;
    float m = 0.f;
    for (int i = blockIdx.x * blockDim.x + threadIdx.x; i < n4; i += stride) {
        if (i + stride < n4) __builtin_prefetch(&x[i + stride], 0, 3);
        float4 v = x[i];
        m = fmaxf(m, fmaxf(fmaxf(fabsf(v.x), fabsf(v.y)),
                           fmaxf(fabsf(v.z), fabsf(v.w))));
    }
    __shared__ float sm[256];
    sm[threadIdx.x] = m;
    __syncthreads();
    for (int off = 128; off; off >>= 1) {
        if ((int)threadIdx.x < off)
            sm[threadIdx.x] = fmaxf(sm[threadIdx.x], sm[threadIdx.x + off]);
        __syncthreads();
    }
    if (threadIdx.x == 0) atomicMax(outBits, __float_as_uint(sm[0]));
}

// ---------------- pass 2: per-row Ex / Ex2 via WMMA IU8 row-sums ---------
// One wave handles 16 rows x 512 cols. Lane L supplies row (L&15); lanes
// 0-15 cover the low 32 cols of each 64-col step, lanes 16-31 the high 32.
// B = all-ones bytes -> D[m][*] = row sum (K-slot permutation invariant).
// The square-LUT map |x_int| -> (sq_hi, sq_lo) is launch-uniform, so it is
// materialized once per block as a 256-entry LDS table indexed by the raw
// signed byte (negatives mirrored) -> 1 ds_load_b32 replaces ~13 VALU ops.
__global__ __launch_bounds__(256) void k_rowstats(const float* __restrict__ x,
                                                  const unsigned* __restrict__ wsu,
                                                  int* __restrict__ exArr,
                                                  int* __restrict__ ex2Arr,
                                                  int N, int chunksPerRow) {
    const int lane     = threadIdx.x & 31;
    const int wid      = blockIdx.x * (blockDim.x >> 5) + (threadIdx.x >> 5);
    const int tile     = wid / chunksPerRow;   // 16-row tile index
    const int chunk    = wid % chunksPerRow;   // 512-col chunk index
    const int rowBase  = tile * 16;
    const int row      = rowBase + (lane & 15);
    const int laneHalf = lane >> 4;

    const float s_in  = scale_from_bits(wsu[0]);
    const float inv_s = 1.0f / s_in;           // uniform; rint(x*inv_s) ~ rint(x/s)

    // ---- per-block square-LUT table: sqtab[b] = (sq_hi << 8) | sq_lo ----
    __shared__ unsigned sqtab[256];
    {
        int t = threadIdx.x;                   // blockDim.x == 256
        int a = (t < 128) ? t : (256 - t);     // |x_int| for signed byte t
        float aq = (float)a * s_in;            // abs_q (real-valued, per ref)
        bool hi = aq >= 64.0f;
        float fh = fminf(floorf(aq * 0.0625f), 15.0f);
        int ih = (int)fh;
        float fm = fmodf(floorf(aq * 0.5f), 16.0f);
        int im = (int)fm;
        unsigned sh = hi ? (unsigned)(ih * ih) : 0u;   // <=225, fits u8
        unsigned sl = hi ? 0u : (unsigned)(im * im);   // <=225, fits u8
        sqtab[t] = (sh << 8) | sl;
    }
    __syncthreads();

    const float* rp = x + (size_t)row * N + chunk * 512 + laneHalf * 32;

    v8i c_ex = {}, c_hi = {}, c_lo = {};
    v8i b1;
#pragma unroll
    for (int j = 0; j < 8; ++j) b1[j] = 0x01010101;

#pragma unroll 2
    for (int it = 0; it < 8; ++it) {  // 8 x 64 cols = 512 cols
        const float4* p = (const float4*)(rp + it * 64);
        v8i a_ex, a_hi, a_lo;
#pragma unroll
        for (int g = 0; g < 8; ++g) {
            float4 f = p[g];
            int pe = 0, ph = 0, pl = 0;
            float fv[4] = {f.x, f.y, f.z, f.w};
#pragma unroll
            for (int j = 0; j < 4; ++j) {
                float r = rintf(fv[j] * inv_s);
                r = fminf(fmaxf(r, -127.f), 127.f);   // x_int (v_med3)
                int vi = (int)r;
                int b  = vi & 255;                    // signed byte of x_int
                unsigned t = sqtab[b];                // (sq_hi<<8)|sq_lo
                pe |= b << (8 * j);
                ph |= ((t >> 8) & 255u) << (8 * j);
                pl |= (t & 255u) << (8 * j);
            }
            a_ex[g] = pe; a_hi[g] = ph; a_lo[g] = pl;
        }
        // signed A (x_int), unsigned ones B  -> row sums of x_int
        c_ex = __builtin_amdgcn_wmma_i32_16x16x64_iu8(true,  a_ex, false, b1, c_ex, false, false);
        // unsigned square-LUT planes
        c_hi = __builtin_amdgcn_wmma_i32_16x16x64_iu8(false, a_hi, false, b1, c_hi, false, false);
        c_lo = __builtin_amdgcn_wmma_i32_16x16x64_iu8(false, a_lo, false, b1, c_lo, false, false);
    }

    // C layout: lanes 0-15 hold M=0..7 (vgpr 0..7), lanes 16-31 hold M=8..15.
    // All columns equal -> lane 0 / lane 16 publish.
    if ((lane & 15) == 0) {
        int rb = rowBase + laneHalf * 8;
#pragma unroll
        for (int r = 0; r < 8; ++r) {
            atomicAdd(&exArr[rb + r], c_ex[r]);
            atomicAdd(&ex2Arr[rb + r], c_hi[r] * 256 + c_lo[r] * 16);
        }
    }
}

// ---------------- pass 2b: per-row mu / inv_std --------------------------
__global__ void k_rowfinal(const unsigned* __restrict__ wsu,
                           const int* __restrict__ exArr,
                           const int* __restrict__ ex2Arr,
                           float* __restrict__ muArr,
                           float* __restrict__ istdArr, int B, int N) {
    int r = blockIdx.x * blockDim.x + threadIdx.x;
    if (r >= B) return;
    float s_in = scale_from_bits(wsu[0]);
    float Ex  = (float)exArr[r] * s_in;      // sum of x_q (exact integer sum)
    float Ex2 = (float)ex2Arr[r];            // sum of sq_decomp (already final units)
    float invN = 1.0f / (float)N;
    float mu  = Ex * invN;
    float var = Ex2 * invN - mu * mu;
    float vint = fminf(fmaxf(rintf(var), 1.0f), 65535.0f);
    int vi = (int)vint;
    int msb = 31 - __clz(vi);                // == floor(log2(var_int))
    muArr[r]   = mu;
    istdArr[r] = c_inv_sqrt_lut[msb] * (1.0f / 65536.0f);
}

// ---------------- shared y computation -----------------------------------
__device__ __forceinline__ float4 compute_y4(float4 xv, float4 g, float4 b,
                                             float s_in, float inv_s,
                                             float mu, float istd) {
    float4 y;
    {
        float r = fminf(fmaxf(rintf(xv.x * inv_s), -127.f), 127.f);
        y.x = (fmaf(r, s_in, -mu)) * (istd * g.x) + b.x;
    }
    {
        float r = fminf(fmaxf(rintf(xv.y * inv_s), -127.f), 127.f);
        y.y = (fmaf(r, s_in, -mu)) * (istd * g.y) + b.y;
    }
    {
        float r = fminf(fmaxf(rintf(xv.z * inv_s), -127.f), 127.f);
        y.z = (fmaf(r, s_in, -mu)) * (istd * g.z) + b.z;
    }
    {
        float r = fminf(fmaxf(rintf(xv.w * inv_s), -127.f), 127.f);
        y.w = (fmaf(r, s_in, -mu)) * (istd * g.w) + b.w;
    }
    return y;
}

// ---------------- pass 3: global max |y| ---------------------------------
__global__ __launch_bounds__(256) void k_normmax(const float4* __restrict__ x,
                                                 const float4* __restrict__ gamma,
                                                 const float4* __restrict__ beta,
                                                 const unsigned* __restrict__ wsu,
                                                 const float* __restrict__ muArr,
                                                 const float* __restrict__ istdArr,
                                                 unsigned* __restrict__ outBits,
                                                 int n4, int rowShift, int rowMask) {
    const float s_in  = scale_from_bits(wsu[0]);
    const float inv_s = 1.0f / s_in;
    const int stride = gridDim.x * blockDim.x;
    float m = 0.f;
    for (int i = blockIdx.x * blockDim.x + threadIdx.x; i < n4; i += stride) {
        if (i + stride < n4) __builtin_prefetch(&x[i + stride], 0, 3);
        int row = i >> rowShift;
        int c4  = i & rowMask;
        float4 y = compute_y4(x[i], gamma[c4], beta[c4], s_in, inv_s,
                              muArr[row], istdArr[row]);
        m = fmaxf(m, fmaxf(fmaxf(fabsf(y.x), fabsf(y.y)),
                           fmaxf(fabsf(y.z), fabsf(y.w))));
    }
    __shared__ float sm[256];
    sm[threadIdx.x] = m;
    __syncthreads();
    for (int off = 128; off; off >>= 1) {
        if ((int)threadIdx.x < off)
            sm[threadIdx.x] = fmaxf(sm[threadIdx.x], sm[threadIdx.x + off]);
        __syncthreads();
    }
    if (threadIdx.x == 0) atomicMax(outBits, __float_as_uint(sm[0]));
}

// ---------------- pass 4: quantized output -------------------------------
__global__ __launch_bounds__(256) void k_output(const float4* __restrict__ x,
                                                const float4* __restrict__ gamma,
                                                const float4* __restrict__ beta,
                                                const unsigned* __restrict__ wsu,
                                                const float* __restrict__ muArr,
                                                const float* __restrict__ istdArr,
                                                float4* __restrict__ out,
                                                int n4, int rowShift, int rowMask) {
    const float s_in    = scale_from_bits(wsu[0]);
    const float inv_s   = 1.0f / s_in;
    const float s_out   = scale_from_bits(wsu[1]);
    const float inv_so  = 1.0f / s_out;
    const int stride = gridDim.x * blockDim.x;
    for (int i = blockIdx.x * blockDim.x + threadIdx.x; i < n4; i += stride) {
        if (i + stride < n4) __builtin_prefetch(&x[i + stride], 0, 3);
        int row = i >> rowShift;
        int c4  = i & rowMask;
        float4 y = compute_y4(x[i], gamma[c4], beta[c4], s_in, inv_s,
                              muArr[row], istdArr[row]);
        float4 o;
        o.x = fminf(fmaxf(rintf(y.x * inv_so), -127.f), 127.f) * s_out;
        o.y = fminf(fmaxf(rintf(y.y * inv_so), -127.f), 127.f) * s_out;
        o.z = fminf(fmaxf(rintf(y.z * inv_so), -127.f), 127.f) * s_out;
        o.w = fminf(fmaxf(rintf(y.w * inv_so), -127.f), 127.f) * s_out;
        out[i] = o;
    }
}

extern "C" void kernel_launch(void* const* d_in, const int* in_sizes, int n_in,
                              void* d_out, int out_size, void* d_ws, size_t ws_size,
                              hipStream_t stream) {
    const float* x     = (const float*)d_in[0];
    const float* gamma = (const float*)d_in[1];
    const float* beta  = (const float*)d_in[2];

    const int N = in_sizes[1];          // 4096
    const int B = in_sizes[0] / N;      // 8192

    // workspace layout (32-bit words):
    // [0] absmax_x bits, [1] absmax_y bits, pad to [8],
    // [8 .. 8+B) Ex_int, [8+B .. 8+2B) Ex2_int, then mu[B], inv_std[B]
    unsigned* wsu   = (unsigned*)d_ws;
    int* exArr      = (int*)d_ws + 8;
    int* ex2Arr     = exArr + B;
    float* muArr    = (float*)(ex2Arr + B);
    float* istdArr  = muArr + B;

    const int initN = 8 + 2 * B;
    k_init<<<(initN + 255) / 256, 256, 0, stream>>>(wsu, initN);

    const int n4 = (int)(((long long)B * N) / 4);
    k_absmax<<<2048, 256, 0, stream>>>((const float4*)x, n4, &wsu[0]);

    const int chunks = N / 512;                 // 8
    const int waves  = (B / 16) * chunks;       // 4096 waves
    k_rowstats<<<waves / 8, 256, 0, stream>>>(x, wsu, exArr, ex2Arr, N, chunks);

    k_rowfinal<<<(B + 255) / 256, 256, 0, stream>>>(wsu, exArr, ex2Arr,
                                                    muArr, istdArr, B, N);

    const int n4PerRow = N / 4;                 // 1024 (power of two)
    int rowShift = 0;
    while ((1 << rowShift) < n4PerRow) ++rowShift;
    const int rowMask = n4PerRow - 1;

    k_normmax<<<2048, 256, 0, stream>>>((const float4*)x, (const float4*)gamma,
                                        (const float4*)beta, wsu, muArr, istdArr,
                                        &wsu[1], n4, rowShift, rowMask);

    k_output<<<2048, 256, 0, stream>>>((const float4*)x, (const float4*)gamma,
                                       (const float4*)beta, wsu, muArr, istdArr,
                                       (float4*)d_out, n4, rowShift, rowMask);
}
